// Translacotron_56169582297703
// MI455X (gfx1250) — matrix operand order repaired and checked by reading, not compiled, not run on hardware
//
#include <hip/hip_runtime.h>
#include <hip/hip_bf16.h>

typedef __bf16 bf16_t;
typedef __attribute__((ext_vector_type(16))) __bf16 v16bf;
typedef __attribute__((ext_vector_type(8)))  float  v8f;

// Problem constants (match reference)
constexpr int Bc   = 16;
constexpr int Tc   = 512;
constexpr int Ec   = 512;
constexpr int Lc   = 128;
constexpr int Vc   = 32000;
constexpr int Dc   = 1024;
constexpr int EMBc = 512;
constexpr int Ac   = 128;
constexpr int Cc   = 10;
constexpr int Kc   = 201;
constexpr float ZPc    = 0.1f;
constexpr float SCALEc = 2.0f;

// ---------------------------------------------------------------------------
// Pack a row-major weight matrix W (Ntot x Ktot, fp32) into WMMA B-fragments
// (bf16). Fragment (nt,kt) = 512 bf16: lane l, element e ->
//   B[k][n] = W[n][k],  k = kt*32 + e + (e>=8 ? 8:0) + (l>=16 ? 8:0),  n = nt*16 + (l&15)
// ---------------------------------------------------------------------------
__global__ void pack_b_kernel(const float* __restrict__ W, bf16_t* __restrict__ dst,
                              int Ktiles, int Ktot, size_t total) {
    size_t idx = (size_t)blockIdx.x * blockDim.x + threadIdx.x;
    if (idx >= total) return;
    int e  = (int)(idx & 15);
    int l  = (int)((idx >> 4) & 31);
    size_t rest = idx >> 9;
    int kt = (int)(rest % Ktiles);
    int nt = (int)(rest / Ktiles);
    int k  = kt * 32 + e + ((e >= 8) ? 8 : 0) + ((l >= 16) ? 8 : 0);
    int n  = nt * 16 + (l & 15);
    dst[idx] = (bf16_t)W[(size_t)n * Ktot + k];
}

// ---------------------------------------------------------------------------
// Pack a row-major activation matrix X (Mtot x Ktot, fp32) into WMMA
// A-fragments (bf16). Fragment (mt,kt): lane l, element e ->
//   m = mt*16 + (l&15),  k = kt*32 + e + (e>=8 ? 8:0) + (l>=16 ? 8:0)
// ---------------------------------------------------------------------------
__global__ void pack_a_kernel(const float* __restrict__ X, bf16_t* __restrict__ dst,
                              int Ktiles, int Ktot, size_t total) {
    size_t idx = (size_t)blockIdx.x * blockDim.x + threadIdx.x;
    if (idx >= total) return;
    int e  = (int)(idx & 15);
    int l  = (int)((idx >> 4) & 31);
    size_t rest = idx >> 9;
    int kt = (int)(rest % Ktiles);
    int mt = (int)(rest / Ktiles);
    int m  = mt * 16 + (l & 15);
    int k  = kt * 32 + e + ((e >= 8) ? 8 : 0) + ((l >= 16) ? 8 : 0);
    dst[idx] = (bf16_t)X[(size_t)m * Ktot + k];
}

// ---------------------------------------------------------------------------
// Generic WMMA GEMM: out(Mtiles*16 x Ntiles*16) = A1*B1 [+ A2*B2] + bias1 [+ bias2]
// One wave per 16x16 output tile; K-loop over bf16 16x16x32 WMMA.
// Fragments are pre-packed (see pack_a/pack_b). out is fp32 row-major with
// row stride `rstride`.
// ---------------------------------------------------------------------------
__global__ void wmma_gemm_kernel(const bf16_t* __restrict__ A1, const bf16_t* __restrict__ B1,
                                 const bf16_t* __restrict__ A2, const bf16_t* __restrict__ B2,
                                 const float* __restrict__ bias1, const float* __restrict__ bias2,
                                 float* __restrict__ out,
                                 int Mtiles, int Ntiles, int Ktiles, long long rstride) {
    int wave = blockIdx.x * (blockDim.x >> 5) + (threadIdx.x >> 5);
    if (wave >= Mtiles * Ntiles) return;           // wave-uniform exit (EXEC all-1 in WMMA)
    int lane = threadIdx.x & 31;
    int mt = wave / Ntiles;
    int nt = wave % Ntiles;

    v8f acc = {};
    const v16bf* Af = (const v16bf*)A1;
    const v16bf* Bf = (const v16bf*)B1;
    for (int kt = 0; kt < Ktiles; ++kt) {
        v16bf a = Af[(size_t)(mt * Ktiles + kt) * 32 + lane];
        v16bf b = Bf[(size_t)(nt * Ktiles + kt) * 32 + lane];
        acc = __builtin_amdgcn_wmma_f32_16x16x32_bf16(false, a, false, b,
                                                      (short)0, acc, false, false);
    }
    if (A2 != nullptr) {
        const v16bf* Af2 = (const v16bf*)A2;
        const v16bf* Bf2 = (const v16bf*)B2;
        for (int kt = 0; kt < Ktiles; ++kt) {
            v16bf a = Af2[(size_t)(mt * Ktiles + kt) * 32 + lane];
            v16bf b = Bf2[(size_t)(nt * Ktiles + kt) * 32 + lane];
            acc = __builtin_amdgcn_wmma_f32_16x16x32_bf16(false, a, false, b,
                                                          (short)0, acc, false, false);
        }
    }

    int half = lane >> 4;
    int col  = lane & 15;
    int n    = nt * 16 + col;
    float badd = 0.f;
    if (bias1) badd += bias1[n];
    if (bias2) badd += bias2[n];
#pragma unroll
    for (int r = 0; r < 8; ++r) {
        int row = mt * 16 + r + 8 * half;
        out[(size_t)row * rstride + n] = acc[r] + badd;
    }
}

// ---------------------------------------------------------------------------
// State / attention-weight init: z0,c0,z1,c1 = 0 ; pw[b,t] = mask/hlens
// ---------------------------------------------------------------------------
__global__ void init_kernel(float* z0, float* c0, float* z1, float* c1,
                            float* pw, const int* __restrict__ hlens) {
    int idx = blockIdx.x * blockDim.x + threadIdx.x;
    if (idx < Bc * Dc) { z0[idx] = 0.f; c0[idx] = 0.f; z1[idx] = 0.f; c1[idx] = 0.f; }
    if (idx < Bc * Tc) {
        int b = idx / Tc, t = idx % Tc;
        int hl = hlens[b];
        pw[idx] = (t < hl) ? (1.0f / (float)hl) : 0.0f;
    }
}

// ---------------------------------------------------------------------------
// Location-aware attention energies: conv1d over prev weights + tanh MLP + gvec dot
// One thread per (b,t). e = -1e30 outside mask.
// ---------------------------------------------------------------------------
__global__ void att_energy_kernel(const float* __restrict__ pw,
                                  const float* __restrict__ encp,   // (B*T, A)
                                  const float* __restrict__ decp,   // (B, A)
                                  const float* __restrict__ locw,   // (C, K)
                                  const float* __restrict__ Watt,   // (A, C)
                                  const float* __restrict__ gv,     // (A)
                                  const int*   __restrict__ hlens,
                                  float* __restrict__ e_out) {
    int idx = blockIdx.x * blockDim.x + threadIdx.x;
    if (idx >= Bc * Tc) return;
    int b = idx / Tc, t = idx % Tc;
    if (t >= hlens[b]) { e_out[idx] = -1e30f; return; }

    float conv[Cc];
#pragma unroll
    for (int c = 0; c < Cc; ++c) conv[c] = 0.f;
    const float* pwb = pw + (size_t)b * Tc;
    for (int k = 0; k < Kc; ++k) {
        int tt = t + k - (Kc / 2);
        if (tt < 0 || tt >= Tc) continue;
        float p = pwb[tt];
#pragma unroll
        for (int c = 0; c < Cc; ++c) conv[c] += p * locw[c * Kc + k];
    }
    const float* ep = encp + (size_t)idx * Ac;
    const float* dp = decp + (size_t)b * Ac;
    float e = 0.f;
    for (int a = 0; a < Ac; ++a) {
        float s = ep[a] + dp[a];
#pragma unroll
        for (int c = 0; c < Cc; ++c) s += conv[c] * Watt[a * Cc + c];
        e += tanhf(s) * gv[a];
    }
    e_out[idx] = e;
}

// ---------------------------------------------------------------------------
// Per-batch softmax (scaled), context vector, and LSTM0 input assembly:
//   w = softmax(2*e); pw <- w; att_c = w @ hs[b]; x = [emb[ys[b,t]], att_c]
// Block b, 512 threads (one per encoder position / feature column).
// ---------------------------------------------------------------------------
__global__ void softmax_attc_kernel(const float* __restrict__ e_in,
                                    const float* __restrict__ hs,   // (B,T,E)
                                    const float* __restrict__ emb,  // (V,EMB)
                                    const int*   __restrict__ ys,   // (B,L)
                                    int tstep,
                                    float* __restrict__ pw,
                                    float* __restrict__ x) {        // (B, EMB+E)
    __shared__ float sh[Tc];
    __shared__ float wsh[Tc];
    int b = blockIdx.x;
    int j = threadIdx.x;

    float ej = SCALEc * e_in[(size_t)b * Tc + j];
    sh[j] = ej;
    __syncthreads();
    for (int s = Tc / 2; s > 0; s >>= 1) {
        if (j < s) sh[j] = fmaxf(sh[j], sh[j + s]);
        __syncthreads();
    }
    float mx = sh[0];
    __syncthreads();
    float ex = expf(ej - mx);
    sh[j] = ex;
    __syncthreads();
    for (int s = Tc / 2; s > 0; s >>= 1) {
        if (j < s) sh[j] += sh[j + s];
        __syncthreads();
    }
    float wj = ex / sh[0];
    wsh[j] = wj;
    pw[(size_t)b * Tc + j] = wj;
    __syncthreads();

    // context column j (E == Tc == 512 so one thread per feature)
    float acc = 0.f;
    const float* hb = hs + (size_t)b * Tc * Ec;
    for (int tt = 0; tt < Tc; ++tt) acc += wsh[tt] * hb[(size_t)tt * Ec + j];

    int y = ys[(size_t)b * Lc + tstep];
    x[(size_t)b * (EMBc + Ec) + j]        = emb[(size_t)y * EMBc + j];
    x[(size_t)b * (EMBc + Ec) + EMBc + j] = acc;
}

// ---------------------------------------------------------------------------
// LSTM cell pointwise + zoneout (eval blend). gates (B, 4D), torch order i,f,g,o.
// In-place update of z, c.
// ---------------------------------------------------------------------------
__device__ __forceinline__ float sigmf(float x) { return 1.0f / (1.0f + expf(-x)); }

__global__ void lstm_pointwise_kernel(const float* __restrict__ gates,
                                      float* __restrict__ z, float* __restrict__ c) {
    int idx = blockIdx.x * blockDim.x + threadIdx.x;
    if (idx >= Bc * Dc) return;
    int b = idx >> 10;          // /1024
    int d = idx & (Dc - 1);
    const float* g = gates + (size_t)b * 4 * Dc;
    float gi = g[d];
    float gf = g[Dc + d];
    float gg = g[2 * Dc + d];
    float go = g[3 * Dc + d];
    float cold = c[idx], zold = z[idx];
    float cn = sigmf(gf) * cold + sigmf(gi) * tanhf(gg);
    float hn = sigmf(go) * tanhf(cn);
    z[idx] = ZPc * zold + (1.0f - ZPc) * hn;
    c[idx] = ZPc * cold + (1.0f - ZPc) * cn;
}

// ---------------------------------------------------------------------------
// Host side
// ---------------------------------------------------------------------------
static inline int cdiv(long long a, long long b) { return (int)((a + b - 1) / b); }

extern "C" void kernel_launch(void* const* d_in, const int* in_sizes, int n_in,
                              void* d_out, int out_size, void* d_ws, size_t ws_size,
                              hipStream_t stream) {
    const float* hs    = (const float*)d_in[0];
    const int*   hlens = (const int*)  d_in[1];
    const int*   ys    = (const int*)  d_in[2];
    const float* emb   = (const float*)d_in[3];
    const float* W_enc = (const float*)d_in[4];
    const float* b_enc = (const float*)d_in[5];
    const float* W_dec = (const float*)d_in[6];
    const float* loc_w = (const float*)d_in[7];
    const float* W_att = (const float*)d_in[8];
    const float* gvec  = (const float*)d_in[9];
    const float* W_ih0 = (const float*)d_in[10]; const float* b_ih0 = (const float*)d_in[11];
    const float* W_hh0 = (const float*)d_in[12]; const float* b_hh0 = (const float*)d_in[13];
    const float* W_ih1 = (const float*)d_in[14]; const float* b_ih1 = (const float*)d_in[15];
    const float* W_hh1 = (const float*)d_in[16]; const float* b_hh1 = (const float*)d_in[17];
    const float* W_out = (const float*)d_in[18]; const float* b_out = (const float*)d_in[19];
    float* out = (float*)d_out;

    // ---- workspace bump allocator (256B aligned) ----
    char* ws = (char*)d_ws;
    size_t off = 0;
    auto alloc = [&](size_t bytes) -> void* {
        off = (off + 255) & ~(size_t)255;
        void* p = ws + off;
        off += bytes;
        return p;
    };

    // Fragment buffers (bf16): fragment = 512 elems
    const int KT_D   = Dc / 32;   // 32  (K=1024)
    const int KT_E   = Ec / 32;   // 16  (K=512)
    const int NT_OUT = Vc / 16;   // 2000
    const int NT_G   = 4 * Dc / 16; // 256
    const int NT_A   = Ac / 16;   // 8
    const int MT_HS  = (Bc * Tc) / 16; // 512

    bf16_t* WoutF = (bf16_t*)alloc((size_t)NT_OUT * KT_D * 512 * 2);
    bf16_t* Wih0F = (bf16_t*)alloc((size_t)NT_G * KT_D * 512 * 2);
    bf16_t* Whh0F = (bf16_t*)alloc((size_t)NT_G * KT_D * 512 * 2);
    bf16_t* Wih1F = (bf16_t*)alloc((size_t)NT_G * KT_D * 512 * 2);
    bf16_t* Whh1F = (bf16_t*)alloc((size_t)NT_G * KT_D * 512 * 2);
    bf16_t* WdecF = (bf16_t*)alloc((size_t)NT_A * KT_D * 512 * 2);
    bf16_t* WencF = (bf16_t*)alloc((size_t)NT_A * KT_E * 512 * 2);
    bf16_t* hsF   = (bf16_t*)alloc((size_t)MT_HS * KT_E * 512 * 2);

    float* encp  = (float*)alloc((size_t)Bc * Tc * Ac * 4);
    float* decp  = (float*)alloc((size_t)Bc * Ac * 4);
    float* evec  = (float*)alloc((size_t)Bc * Tc * 4);
    float* pw    = (float*)alloc((size_t)Bc * Tc * 4);
    float* xbuf  = (float*)alloc((size_t)Bc * (EMBc + Ec) * 4);
    float* gates = (float*)alloc((size_t)Bc * 4 * Dc * 4);
    float* z0    = (float*)alloc((size_t)Bc * Dc * 4);
    float* c0    = (float*)alloc((size_t)Bc * Dc * 4);
    float* z1    = (float*)alloc((size_t)Bc * Dc * 4);
    float* c1    = (float*)alloc((size_t)Bc * Dc * 4);
    bf16_t* aX   = (bf16_t*)alloc((size_t)KT_D * 512 * 2);
    bf16_t* aH   = (bf16_t*)alloc((size_t)KT_D * 512 * 2);
    (void)ws_size; (void)in_sizes; (void)n_in; (void)out_size;

    const int PK = 256;                    // pack-kernel block size
    const int GB = 128;                    // gemm block size (4 waves)
    auto gemm_blocks = [](int waves) { return (waves + 3) / 4; };

    // ---- one-time (per call) weight packing + encoder projection ----
    {
        size_t tot;
        tot = (size_t)NT_OUT * KT_D * 512;
        pack_b_kernel<<<cdiv(tot, PK), PK, 0, stream>>>(W_out, WoutF, KT_D, Dc, tot);
        tot = (size_t)NT_G * KT_D * 512;
        pack_b_kernel<<<cdiv(tot, PK), PK, 0, stream>>>(W_ih0, Wih0F, KT_D, Dc, tot);
        pack_b_kernel<<<cdiv(tot, PK), PK, 0, stream>>>(W_hh0, Whh0F, KT_D, Dc, tot);
        pack_b_kernel<<<cdiv(tot, PK), PK, 0, stream>>>(W_ih1, Wih1F, KT_D, Dc, tot);
        pack_b_kernel<<<cdiv(tot, PK), PK, 0, stream>>>(W_hh1, Whh1F, KT_D, Dc, tot);
        tot = (size_t)NT_A * KT_D * 512;
        pack_b_kernel<<<cdiv(tot, PK), PK, 0, stream>>>(W_dec, WdecF, KT_D, Dc, tot);
        tot = (size_t)NT_A * KT_E * 512;
        pack_b_kernel<<<cdiv(tot, PK), PK, 0, stream>>>(W_enc, WencF, KT_E, Ec, tot);
        tot = (size_t)MT_HS * KT_E * 512;
        pack_a_kernel<<<cdiv(tot, PK), PK, 0, stream>>>(hs, hsF, KT_E, Ec, tot);

        // enc_proj = hs @ W_enc^T + b_enc   (M=8192, N=128, K=512)
        int waves = MT_HS * NT_A;
        wmma_gemm_kernel<<<gemm_blocks(waves), GB, 0, stream>>>(
            hsF, WencF, nullptr, nullptr, b_enc, nullptr,
            encp, MT_HS, NT_A, KT_E, (long long)Ac);

        init_kernel<<<cdiv(Bc * Dc, PK), PK, 0, stream>>>(z0, c0, z1, c1, pw, hlens);
    }

    // ---- 128 sequential decoder steps ----
    for (int t = 0; t < Lc; ++t) {
        size_t totA = (size_t)KT_D * 512;

        // pack z0 (carry) -> aH : used for dec-state projection and LSTM0 h-input
        pack_a_kernel<<<cdiv(totA, PK), PK, 0, stream>>>(z0, aH, KT_D, Dc, totA);

        // decp = z0 @ W_dec^T     (M=16, N=128, K=1024)
        wmma_gemm_kernel<<<gemm_blocks(NT_A), GB, 0, stream>>>(
            aH, WdecF, nullptr, nullptr, nullptr, nullptr,
            decp, 1, NT_A, KT_D, (long long)Ac);

        // attention energies + masked fill
        att_energy_kernel<<<cdiv(Bc * Tc, PK), PK, 0, stream>>>(
            pw, encp, decp, loc_w, W_att, gvec, hlens, evec);

        // softmax + context + x = [emb_t, att_c]; updates pw
        softmax_attc_kernel<<<Bc, Tc, 0, stream>>>(evec, hs, emb, ys, t, pw, xbuf);

        // LSTM0 gates = x @ Wih0^T + z0 @ Whh0^T + b_ih0 + b_hh0  (N=4096, K=1024 each)
        pack_a_kernel<<<cdiv(totA, PK), PK, 0, stream>>>(xbuf, aX, KT_D, Dc, totA);
        wmma_gemm_kernel<<<gemm_blocks(NT_G), GB, 0, stream>>>(
            aX, Wih0F, aH, Whh0F, b_ih0, b_hh0,
            gates, 1, NT_G, KT_D, (long long)(4 * Dc));
        lstm_pointwise_kernel<<<cdiv(Bc * Dc, PK), PK, 0, stream>>>(gates, z0, c0);

        // LSTM1 gates = z0_new @ Wih1^T + z1 @ Whh1^T + b_ih1 + b_hh1
        pack_a_kernel<<<cdiv(totA, PK), PK, 0, stream>>>(z0, aX, KT_D, Dc, totA);
        pack_a_kernel<<<cdiv(totA, PK), PK, 0, stream>>>(z1, aH, KT_D, Dc, totA);
        wmma_gemm_kernel<<<gemm_blocks(NT_G), GB, 0, stream>>>(
            aX, Wih1F, aH, Whh1F, b_ih1, b_hh1,
            gates, 1, NT_G, KT_D, (long long)(4 * Dc));
        lstm_pointwise_kernel<<<cdiv(Bc * Dc, PK), PK, 0, stream>>>(gates, z1, c1);

        // out[:, t, :] = z1_new @ W_out^T + b_out   (M=16, N=32000, K=1024)
        pack_a_kernel<<<cdiv(totA, PK), PK, 0, stream>>>(z1, aX, KT_D, Dc, totA);
        wmma_gemm_kernel<<<gemm_blocks(NT_OUT), GB, 0, stream>>>(
            aX, WoutF, nullptr, nullptr, b_out, nullptr,
            out + (size_t)t * Vc, 1, NT_OUT, KT_D, (long long)Lc * Vc);
    }
}
